// MLPDecoder_11605001634488
// MI455X (gfx1250) — matrix operand order repaired and checked by reading, not compiled, not run on hardware
//
#include <hip/hip_runtime.h>
#include <hip/hip_bf16.h>
#include <stdint.h>

// ---------------- problem constants (from reference) ----------------
#define BATCH    32
#define NATOMS   2000
#define TDIM     128            // n_in
#define MSG_HID  256
#define MSG_OUT  128
#define N_HID    256
#define N_CLS    10
#define ETYPES   2
#define DEG      16
#define BN       (BATCH * NATOMS)          // 64000
#define NEDGE    (BN * DEG)                // 1,024,000 per type

typedef __attribute__((ext_vector_type(16))) __bf16 v16bf;
typedef __attribute__((ext_vector_type(8)))  __bf16 v8bf;
typedef __attribute__((ext_vector_type(8)))  float  v8f;

// ---------------- utility: zero a float buffer ----------------
__global__ void zero_f32(float* __restrict__ p, long n) {
    long i = (long)blockIdx.x * blockDim.x + threadIdx.x;
    long stride = (long)gridDim.x * blockDim.x;
    for (; i < n; i += stride) p[i] = 0.0f;
}

// ---------------- scatter-add: dst[recv[e]] += src[send[e]], rows of 128 f32
// one wave32 per edge; each lane moves one float4 (32*4 = 128 floats)
__global__ void scatter_add_128(float* __restrict__ dst,
                                const float* __restrict__ src,
                                const int* __restrict__ send,
                                const int* __restrict__ recv,
                                int nEdges) {
    int e = blockIdx.x * (blockDim.x >> 5) + (threadIdx.x >> 5);
    if (e >= nEdges) return;
    int lane = threadIdx.x & 31;
    int s = send[e];
    int r = recv[e];
    const float4 v = ((const float4*)(src + (size_t)s * 128))[lane];
    float* dp = dst + (size_t)r * 128 + lane * 4;
    atomicAdd(dp + 0, v.x);   // global_atomic_add_f32, no-return, L2-resident
    atomicAdd(dp + 1, v.y);
    atomicAdd(dp + 2, v.z);
    atomicAdd(dp + 3, v.w);
}

// ---------------- WMMA bf16 GEMM: C = [relu](A[M,K] @ B[K,N] + bias)
// block = 256 threads (8 waves), block tile M=128 x N=64, K-step 32.
// A tile staged f32 via async global->LDS (ASYNCcnt path), converted to bf16
// at fragment build; B tile staged synchronously with transpose as bf16.
// Requires: M % 128 == 0, N % 64 == 0, K % 32 == 0 (true for all calls).
#define STRA32 36  // f32 stride of A rows in LDS (32 + pad), keeps 16B chunks aligned
#define STRB   40  // bf16 stride of B^T rows in LDS

__global__ void gemm_bias_relu_wmma(const float* __restrict__ A,
                                    const float* __restrict__ Bm,
                                    const float* __restrict__ bias,   // may be null
                                    float* __restrict__ C,
                                    int M, int K, int Ntot, int doRelu) {
    __shared__ float  lds_a32[128 * STRA32];   // A tile: 128 rows x 32 k (f32)
    __shared__ __bf16 lds_bt[64 * STRB];       // B tile transposed: 64 cols x 32 k

    const int tid  = threadIdx.x;
    const int wave = tid >> 5;
    const int lane = tid & 31;
    const int m0 = blockIdx.x * 128;
    const int n0 = blockIdx.y * 64;

    const int half = lane >> 4;   // 0: lanes 0-15, 1: lanes 16-31
    const int l16  = lane & 15;

    v8f acc0 = {}, acc1 = {}, acc2 = {}, acc3 = {};
    v8f accs[4] = {acc0, acc1, acc2, acc3};

    for (int k0 = 0; k0 < K; k0 += 32) {
        // ---- async-stage A tile: 128 rows x 32 f32 = 1024 chunks of 16B ----
        #pragma unroll
        for (int it = 0; it < 4; it++) {
            int idx   = tid + it * 256;         // 0..1023
            int row   = idx >> 3;
            int chunk = idx & 7;                // 8 x 16B per row
            unsigned ldsoff = (unsigned)(uintptr_t)(lds_a32 + row * STRA32 + chunk * 4);
            const float* gp = A + (size_t)(m0 + row) * K + (k0 + chunk * 4);
            asm volatile("global_load_async_to_lds_b128 %0, %1, off"
                         :: "v"(ldsoff), "v"(gp) : "memory");
        }
        // ---- stage B tile transposed: lds_bt[n][k] (coalesced over n) ----
        for (int idx = tid; idx < 32 * 64; idx += 256) {
            int k = idx >> 6, n = idx & 63;
            lds_bt[n * STRB + k] = (__bf16)Bm[(size_t)(k0 + k) * Ntot + (n0 + n)];
        }
        asm volatile("s_wait_asynccnt 0x0" ::: "memory");
        __syncthreads();

        // A fragment: row = wave*16 + l16; k = half*8 + {0..7}, half*8 + 16 + {0..7}
        const int arow = wave * 16 + l16;
        const int kb = half * 8;
        const float* ar = lds_a32 + arow * STRA32 + kb;
        float4 a0 = *(const float4*)(ar);
        float4 a1 = *(const float4*)(ar + 4);
        float4 a2 = *(const float4*)(ar + 16);
        float4 a3 = *(const float4*)(ar + 20);
        v16bf afrag;
        afrag[0]  = (__bf16)a0.x; afrag[1]  = (__bf16)a0.y;
        afrag[2]  = (__bf16)a0.z; afrag[3]  = (__bf16)a0.w;
        afrag[4]  = (__bf16)a1.x; afrag[5]  = (__bf16)a1.y;
        afrag[6]  = (__bf16)a1.z; afrag[7]  = (__bf16)a1.w;
        afrag[8]  = (__bf16)a2.x; afrag[9]  = (__bf16)a2.y;
        afrag[10] = (__bf16)a2.z; afrag[11] = (__bf16)a2.w;
        afrag[12] = (__bf16)a3.x; afrag[13] = (__bf16)a3.y;
        afrag[14] = (__bf16)a3.z; afrag[15] = (__bf16)a3.w;

        // B fragments: col = nf*16 + l16; k = half*8 + {0..7}, half*8 + 16 + {0..7}
        #pragma unroll
        for (int nf = 0; nf < 4; nf++) {
            const int bcol = nf * 16 + l16;
            v8bf blo = *(const v8bf*)(lds_bt + bcol * STRB + half * 8);
            v8bf bhi = *(const v8bf*)(lds_bt + bcol * STRB + half * 8 + 16);
            v16bf bfrag = __builtin_shufflevector(blo, bhi,
                             0,1,2,3,4,5,6,7,8,9,10,11,12,13,14,15);
            accs[nf] = __builtin_amdgcn_wmma_f32_16x16x32_bf16(
                false, afrag, false, bfrag, (short)0, accs[nf], false, false);
        }
        __syncthreads();
    }

    // epilogue: C/D layout — VGPR v: lanes 0-15 -> M=v, lanes 16-31 -> M=v+8; N=l16
    #pragma unroll
    for (int nf = 0; nf < 4; nf++) {
        const int col = n0 + nf * 16 + l16;
        const float bv = bias ? bias[col] : 0.0f;
        #pragma unroll
        for (int v = 0; v < 8; v++) {
            const int row = m0 + wave * 16 + half * 8 + v;
            float x = accs[nf][v] + bv;
            if (doRelu) x = fmaxf(x, 0.0f);
            C[(size_t)row * Ntot + col] = x;
        }
    }
}

// ---------------- allagg += relu(tmp + bias[col & 127]), rows of 128
__global__ void acc_bias_relu_128(float* __restrict__ allagg,
                                  const float* __restrict__ tmp,
                                  const float* __restrict__ bias,
                                  long n) {
    long i = (long)blockIdx.x * blockDim.x + threadIdx.x;
    long stride = (long)gridDim.x * blockDim.x;
    for (; i < n; i += stride) {
        float x = tmp[i] + bias[(int)(i & 127)];
        allagg[i] += fmaxf(x, 0.0f);
    }
}

// ---------------- partial mean over atoms: mean[b,f] += (1/N)*sum_rows pred
// grid (BATCH, chunks), block 256 (one thread per feature)
__global__ void mean_partial(float* __restrict__ mean,
                             const float* __restrict__ pred,
                             int rowsPerChunk) {
    const int b = blockIdx.x;
    const int n0 = blockIdx.y * rowsPerChunk;
    const int f = threadIdx.x;
    float s = 0.0f;
    for (int n = 0; n < rowsPerChunk; n++)
        s += pred[((size_t)b * NATOMS + n0 + n) * N_HID + f];
    atomicAdd(&mean[b * N_HID + f], s * (1.0f / (float)NATOMS));
}

// ---------------- final tiny GEMM: out[32,10] = mean @ Wo3 + bo3
__global__ void final_proj(float* __restrict__ out,
                           const float* __restrict__ mean,
                           const float* __restrict__ Wo3,
                           const float* __restrict__ bo3) {
    int idx = blockIdx.x * blockDim.x + threadIdx.x;
    if (idx >= BATCH * N_CLS) return;
    int b = idx / N_CLS, c = idx % N_CLS;
    float s = bo3[c];
    for (int f = 0; f < N_HID; f++)
        s += mean[b * N_HID + f] * Wo3[f * N_CLS + c];
    out[idx] = s;
}

// ---------------- launcher ----------------
extern "C" void kernel_launch(void* const* d_in, const int* in_sizes, int n_in,
                              void* d_out, int out_size, void* d_ws, size_t ws_size,
                              hipStream_t stream) {
    const float* X    = (const float*)d_in[0];   // [B,N,T] -> [BN,128]
    const float* W1   = (const float*)d_in[1];   // [2,128,256]
    const float* b1   = (const float*)d_in[2];   // [2,256]
    const float* W2   = (const float*)d_in[3];   // [2,256,128]
    const float* b2   = (const float*)d_in[4];   // [2,128]
    const float* Wo1  = (const float*)d_in[5];   // [128,256]
    const float* bo1  = (const float*)d_in[6];   // [256]
    const float* Wo2  = (const float*)d_in[7];   // [256,256]
    const float* bo2  = (const float*)d_in[8];   // [256]
    const float* Wo3  = (const float*)d_in[9];   // [256,10]
    const float* bo3  = (const float*)d_in[10];  // [10]
    const int* esend  = (const int*)d_in[11];    // [2,E]
    const int* erecv  = (const int*)d_in[12];    // [2,E]
    float* out        = (float*)d_out;           // [32,10]

    // workspace partition (floats); total ~188 MiB
    float* ws      = (float*)d_ws;
    float* allagg  = ws;                                   // BN*128
    float* bufA    = allagg + (size_t)BN * 128;            // BN*128 (agg, then h@W2)
    float* bufB    = bufA   + (size_t)BN * 128;            // BN*256 (h, then pred1)
    float* bufC    = bufB   + (size_t)BN * 256;            // BN*256 (tmp128, then pred2)
    float* meanb   = bufC   + (size_t)BN * 256;            // BATCH*256

    const dim3 blk256(256);
    const dim3 zgrid(2048);
    const int scatterBlocks = NEDGE / 8;                   // 8 edges (waves) per block

    // allagg = 0
    zero_f32<<<zgrid, blk256, 0, stream>>>(allagg, (long)BN * 128);

    for (int i = 0; i < ETYPES; i++) {
        const int*   es  = esend + (size_t)i * NEDGE;
        const int*   er  = erecv + (size_t)i * NEDGE;
        const float* W1i = W1 + (size_t)i * TDIM * MSG_HID;
        const float* b1i = b1 + (size_t)i * MSG_HID;
        const float* W2i = W2 + (size_t)i * MSG_HID * MSG_OUT;
        const float* b2i = b2 + (size_t)i * MSG_OUT;

        // agg = segment_sum(X[s], r)           [BN,128]
        zero_f32<<<zgrid, blk256, 0, stream>>>(bufA, (long)BN * 128);
        scatter_add_128<<<scatterBlocks, blk256, 0, stream>>>(bufA, X, es, er, NEDGE);

        // h = relu(agg @ W1_i + b1_i)          [BN,256]
        gemm_bias_relu_wmma<<<dim3(BN / 128, MSG_HID / 64), blk256, 0, stream>>>(
            bufA, W1i, b1i, bufB, BN, TDIM, MSG_HID, 1);

        // h2 = h @ W2_i  (project BEFORE the segment-sum: linearity halves gather)
        gemm_bias_relu_wmma<<<dim3(BN / 128, MSG_OUT / 64), blk256, 0, stream>>>(
            bufB, W2i, nullptr, bufA, BN, MSG_HID, MSG_OUT, 0);

        // tmp = segment_sum(h2[s], r)          [BN,128]
        zero_f32<<<zgrid, blk256, 0, stream>>>(bufC, (long)BN * 128);
        scatter_add_128<<<scatterBlocks, blk256, 0, stream>>>(bufC, bufA, es, er, NEDGE);

        // allagg += relu(tmp + b2_i)
        acc_bias_relu_128<<<zgrid, blk256, 0, stream>>>(allagg, bufC, b2i, (long)BN * 128);
    }

    // pred1 = relu(allagg @ Wo1 + bo1)         [BN,256]
    gemm_bias_relu_wmma<<<dim3(BN / 128, N_HID / 64), blk256, 0, stream>>>(
        allagg, Wo1, bo1, bufB, BN, MSG_OUT, N_HID, 1);

    // pred2 = relu(pred1 @ Wo2 + bo2)          [BN,256]
    gemm_bias_relu_wmma<<<dim3(BN / 128, N_HID / 64), blk256, 0, stream>>>(
        bufB, Wo2, bo2, bufC, BN, N_HID, N_HID, 1);

    // mean over atoms then final projection
    zero_f32<<<dim3(32), blk256, 0, stream>>>(meanb, (long)BATCH * N_HID);
    mean_partial<<<dim3(BATCH, 8), blk256, 0, stream>>>(meanb, bufC, NATOMS / 8);
    final_proj<<<dim3(2), blk256, 0, stream>>>(out, meanb, Wo3, bo3);
}
